// ResidualBlockGAT_27599459844198
// MI455X (gfx1250) — matrix operand order repaired and checked by reading, not compiled, not run on hardware
//
#include <hip/hip_runtime.h>
#include <math.h>

#define HEADS 4
#define CH 16
#define ODIM 64
#define IDIM 128
#define GG 64
#define EPSN 1e-5f
#define SLOPE 0.2f

typedef unsigned short us;
typedef __attribute__((ext_vector_type(16))) __bf16 v16bf;
typedef __attribute__((ext_vector_type(8)))  float v8f;

union BFrag { uint4 q[2]; v16bf v; };

__device__ __forceinline__ us f2bf(float f) {
    union { float f; unsigned u; } x; x.f = f;
    unsigned u = x.u;
    unsigned r = u + 0x7FFFu + ((u >> 16) & 1u);   // round-to-nearest-even
    if ((u & 0x7F800000u) == 0x7F800000u) r = u;   // inf/nan passthrough
    return (us)(r >> 16);
}

__device__ __forceinline__ void atomicMaxF(float* addr, float v) {
    if (v >= 0.f) atomicMax((int*)addr, __float_as_int(v));
    else          atomicMin((unsigned int*)addr, __float_as_uint(v));
}

// ---------------- init ----------------
__global__ __launch_bounds__(256) void k_init(float* accum, float* smax, float* denom,
                                              float* gstat, float* cnt, int N) {
    int i = blockIdx.x * blockDim.x + threadIdx.x;
    if (i < N * ODIM)        accum[i] = 0.f;
    if (i < N * HEADS)     { smax[i] = -INFINITY; denom[i] = 0.f; }
    if (i < 4 * GG * ODIM)   gstat[i] = 0.f;
    if (i < GG)              cnt[i] = 0.f;
}

// ---------------- f32 -> bf16 converts ----------------
__global__ __launch_bounds__(256) void k_cvt_x(const float* __restrict__ x, us* __restrict__ xb, int n) {
    int i = blockIdx.x * blockDim.x + threadIdx.x;
    if (i < n) xb[i] = f2bf(x[i]);
}

__global__ __launch_bounds__(256) void k_cvt_w(const float* __restrict__ wl, const float* __restrict__ wr,
                                               const float* __restrict__ wsk, us* __restrict__ wb) {
    int i = blockIdx.x * blockDim.x + threadIdx.x;
    const int WSZ = IDIM * ODIM;
    if (i >= 3 * WSZ) return;
    int j = i % WSZ, src = (j % IDIM) * ODIM + j / IDIM;
    float v = (i < WSZ) ? wl[src] : (i < 2 * WSZ ? wr[src] : wsk[src]);
    wb[i] = f2bf(v);
}

// ---------------- fused 3-weight WMMA GEMM: xl = x@Wl, xr = x@Wr, xsk = x@Ws + b ----------------
__global__ __launch_bounds__(256) void k_gemm(const us* __restrict__ xb, const us* __restrict__ wb,
                                              const float* __restrict__ skip_b,
                                              float* __restrict__ xl, float* __restrict__ xr,
                                              float* __restrict__ xsk, int N) {
    int wave = threadIdx.x >> 5;
    int lane = threadIdx.x & 31;
    int base = (blockIdx.x * 8 + wave) * 16;       // 16-row node tile per wave
    if (base >= N) return;                          // wave-uniform (EXEC all-1 for WMMA)
    int m  = lane & 15;
    int kh = lane >> 4;

    // Preload A fragments: lane holds row (base+m), K elems {kh*8..+7, 16+kh*8..+7} per 32-K step
    BFrag a[4];
    const uint4* xq = (const uint4*)(xb + (size_t)(base + m) * IDIM);
#pragma unroll
    for (int s = 0; s < 4; ++s) {
        a[s].q[0] = xq[4 * s + kh];
        a[s].q[1] = xq[4 * s + kh + 2];
    }

    float* outs[3] = { xl, xr, xsk };
#pragma unroll
    for (int w = 0; w < 3; ++w) {
        const us* wp = wb + (size_t)w * (IDIM * ODIM);
#pragma unroll
        for (int t = 0; t < 4; ++t) {
            v8f acc = {0.f, 0.f, 0.f, 0.f, 0.f, 0.f, 0.f, 0.f};
#pragma unroll
            for (int s = 0; s < 4; ++s) {
                // B fragment: lane = N col (16t+m), K elems {kh*8..+7, 16+kh*8..+7} (W stored [n][k])
                BFrag b;
                const uint4* bq = (const uint4*)(wp + (size_t)(16 * t + m) * IDIM + 32 * s);
                b.q[0] = bq[kh];
                b.q[1] = bq[kh + 2];
                acc = __builtin_amdgcn_wmma_f32_16x16x32_bf16(
                        false, a[s].v, false, b.v, (short)0, acc, false, false);
            }
            int col = 16 * t + m;
            float bias = (w == 2) ? skip_b[col] : 0.f;
            float* op = outs[w] + (size_t)(base + kh * 8) * ODIM + col;
#pragma unroll
            for (int v = 0; v < 8; ++v)
                op[(size_t)v * ODIM] = acc[v] + bias;   // D: VGPR v -> row v (+8 for hi lanes)
        }
    }
}

// ---------------- edge pass 1: GATv2 scores + segment max ----------------
__global__ __launch_bounds__(256) void k_edge_score(const int* __restrict__ ei,
                                                    const float* __restrict__ xl, const float* __restrict__ xr,
                                                    const float* __restrict__ att,
                                                    float* __restrict__ score, float* smax,
                                                    int E, int Etot) {
    int tid = blockIdx.x * blockDim.x + threadIdx.x;
    if (tid >= Etot * HEADS) return;
    int e = tid >> 2, h = tid & 3;
    int s, d;
    if (e < E) { s = ei[e]; d = ei[E + e]; } else { s = d = e - E; }
    const float4* pl = (const float4*)(xl + (size_t)s * ODIM + h * CH);
    const float4* pr = (const float4*)(xr + (size_t)d * ODIM + h * CH);
    const float4* pa = (const float4*)(att + h * CH);
    float sc = 0.f;
#pragma unroll
    for (int i = 0; i < 4; ++i) {
        float4 A = pl[i], B = pr[i], W = pa[i];
        float t0 = A.x + B.x; t0 = t0 > 0.f ? t0 : SLOPE * t0;
        float t1 = A.y + B.y; t1 = t1 > 0.f ? t1 : SLOPE * t1;
        float t2 = A.z + B.z; t2 = t2 > 0.f ? t2 : SLOPE * t2;
        float t3 = A.w + B.w; t3 = t3 > 0.f ? t3 : SLOPE * t3;
        sc += W.x * t0 + W.y * t1 + W.z * t2 + W.w * t3;
    }
    score[(size_t)e * HEADS + h] = sc;
    atomicMaxF(&smax[(size_t)d * HEADS + h], sc);
}

// ---------------- edge pass 2: exp + segment sum ----------------
__global__ __launch_bounds__(256) void k_edge_exp(const int* __restrict__ ei,
                                                  float* __restrict__ score,
                                                  const float* __restrict__ smax,
                                                  float* denom, int E, int Etot) {
    int tid = blockIdx.x * blockDim.x + threadIdx.x;
    if (tid >= Etot * HEADS) return;
    int e = tid >> 2, h = tid & 3;
    int d = (e < E) ? ei[E + e] : (e - E);
    size_t idx = (size_t)e * HEADS + h;
    float ex = __expf(score[idx] - smax[(size_t)d * HEADS + h]);
    score[idx] = ex;
    unsafeAtomicAdd(&denom[(size_t)d * HEADS + h], ex);
}

// ---------------- edge pass 3: alpha-weighted message scatter ----------------
__global__ __launch_bounds__(256) void k_edge_scatter(const int* __restrict__ ei,
                                                      const float* __restrict__ score,
                                                      const float* __restrict__ denom,
                                                      const float* __restrict__ xl,
                                                      float* accum, int E, int Etot) {
    int tid = blockIdx.x * blockDim.x + threadIdx.x;
    if (tid >= Etot * HEADS) return;
    int e = tid >> 2, h = tid & 3;
    int s, d;
    if (e < E) { s = ei[e]; d = ei[E + e]; } else { s = d = e - E; }
    float alpha = score[(size_t)e * HEADS + h] / denom[(size_t)d * HEADS + h];
    const float4* pl = (const float4*)(xl + (size_t)s * ODIM + h * CH);
    float* op = accum + (size_t)d * ODIM + h * CH;
#pragma unroll
    for (int i = 0; i < 4; ++i) {
        float4 A = pl[i];
        unsafeAtomicAdd(op + 4 * i + 0, A.x * alpha);
        unsafeAtomicAdd(op + 4 * i + 1, A.y * alpha);
        unsafeAtomicAdd(op + 4 * i + 2, A.z * alpha);
        unsafeAtomicAdd(op + 4 * i + 3, A.w * alpha);
    }
}

// ---------------- GraphNorm pass 1: per-graph sums + counts ----------------
__global__ __launch_bounds__(256) void k_gsum(const float* __restrict__ accum, const float* __restrict__ xsk,
                                              const float* __restrict__ conv_bias,
                                              const int* __restrict__ batch,
                                              float* gstat, float* cnt, int N) {
    int i = blockIdx.x * blockDim.x + threadIdx.x;
    if (i >= N * ODIM) return;
    int node = i >> 6, c = i & 63;
    int g = batch[node];
    unsafeAtomicAdd(&gstat[(size_t)g * ODIM + c], accum[i] + conv_bias[c]);
    unsafeAtomicAdd(&gstat[(size_t)(GG + g) * ODIM + c], xsk[i]);
    if (c == 0) unsafeAtomicAdd(&cnt[g], 1.f);
}

// ---------------- GraphNorm pass 2: centered sum of squares ----------------
__global__ __launch_bounds__(256) void k_gvar(const float* __restrict__ accum, const float* __restrict__ xsk,
                                              const float* __restrict__ conv_bias,
                                              const int* __restrict__ batch,
                                              const float* __restrict__ bn_ms, const float* __restrict__ sn_ms,
                                              float* gstat, const float* __restrict__ cnt, int N) {
    int i = blockIdx.x * blockDim.x + threadIdx.x;
    if (i >= N * ODIM) return;
    int node = i >> 6, c = i & 63;
    int g = batch[node];
    float cn = fmaxf(cnt[g], 1.f);
    float mean_m = gstat[(size_t)g * ODIM + c] / cn;
    float mean_s = gstat[(size_t)(GG + g) * ODIM + c] / cn;
    float cm = accum[i] + conv_bias[c] - mean_m * bn_ms[c];
    float cs = xsk[i] - mean_s * sn_ms[c];
    unsafeAtomicAdd(&gstat[(size_t)(2 * GG + g) * ODIM + c], cm * cm);
    unsafeAtomicAdd(&gstat[(size_t)(3 * GG + g) * ODIM + c], cs * cs);
}

// ---------------- final: normalize both branches, add, ELU ----------------
__global__ __launch_bounds__(256) void k_final(const float* __restrict__ accum, const float* __restrict__ xsk,
                                               const float* __restrict__ conv_bias,
                                               const int* __restrict__ batch,
                                               const float* __restrict__ bn_w, const float* __restrict__ bn_b,
                                               const float* __restrict__ bn_ms,
                                               const float* __restrict__ sn_w, const float* __restrict__ sn_b,
                                               const float* __restrict__ sn_ms,
                                               const float* __restrict__ gstat, const float* __restrict__ cnt,
                                               float* __restrict__ out, int N) {
    int i = blockIdx.x * blockDim.x + threadIdx.x;
    if (i >= N * ODIM) return;
    int node = i >> 6, c = i & 63;
    int g = batch[node];
    float cn = fmaxf(cnt[g], 1.f);
    float mean_m = gstat[(size_t)g * ODIM + c] / cn;
    float mean_s = gstat[(size_t)(GG + g) * ODIM + c] / cn;
    float var_m = gstat[(size_t)(2 * GG + g) * ODIM + c] / cn;
    float var_s = gstat[(size_t)(3 * GG + g) * ODIM + c] / cn;
    float inv_m = rsqrtf(var_m + EPSN);
    float inv_s = rsqrtf(var_s + EPSN);
    float cm = accum[i] + conv_bias[c] - mean_m * bn_ms[c];
    float cs = xsk[i] - mean_s * sn_ms[c];
    float y = bn_w[c] * cm * inv_m + bn_b[c] + sn_w[c] * cs * inv_s + sn_b[c];
    out[i] = y > 0.f ? y : (__expf(y) - 1.f);
}

extern "C" void kernel_launch(void* const* d_in, const int* in_sizes, int n_in,
                              void* d_out, int out_size, void* d_ws, size_t ws_size,
                              hipStream_t stream) {
    const float* x        = (const float*)d_in[0];
    const int*   ei       = (const int*)d_in[1];
    const int*   batch    = (const int*)d_in[2];
    const float* Wl       = (const float*)d_in[3];
    const float* Wr       = (const float*)d_in[4];
    const float* att      = (const float*)d_in[5];
    const float* conv_b   = (const float*)d_in[6];
    const float* skip_W   = (const float*)d_in[7];
    const float* skip_b   = (const float*)d_in[8];
    const float* bn_w     = (const float*)d_in[9];
    const float* bn_b     = (const float*)d_in[10];
    const float* bn_ms    = (const float*)d_in[11];
    const float* sn_w     = (const float*)d_in[12];
    const float* sn_b     = (const float*)d_in[13];
    const float* sn_ms    = (const float*)d_in[14];

    const int N    = in_sizes[0] / IDIM;
    const int E    = in_sizes[1] / 2;
    const int Etot = E + N;

    // workspace carve-up (256B aligned)
    char* p = (char*)d_ws;
    auto carve = [&](size_t bytes) { void* r = (void*)p; p += (bytes + 255) & ~(size_t)255; return r; };
    us*    xb    = (us*)carve((size_t)N * IDIM * sizeof(us));
    us*    wb    = (us*)carve((size_t)3 * IDIM * ODIM * sizeof(us));
    float* xl    = (float*)carve((size_t)N * ODIM * sizeof(float));
    float* xr    = (float*)carve((size_t)N * ODIM * sizeof(float));
    float* xsk   = (float*)carve((size_t)N * ODIM * sizeof(float));
    float* accum = (float*)carve((size_t)N * ODIM * sizeof(float));
    float* score = (float*)carve((size_t)Etot * HEADS * sizeof(float));
    float* smax  = (float*)carve((size_t)N * HEADS * sizeof(float));
    float* denom = (float*)carve((size_t)N * HEADS * sizeof(float));
    float* gstat = (float*)carve((size_t)4 * GG * ODIM * sizeof(float));
    float* cnt   = (float*)carve((size_t)GG * sizeof(float));

    const int TB = 256;
    int nNodeElems = N * ODIM;

    k_init<<<(nNodeElems + TB - 1) / TB, TB, 0, stream>>>(accum, smax, denom, gstat, cnt, N);
    k_cvt_x<<<((size_t)N * IDIM + TB - 1) / TB, TB, 0, stream>>>(x, xb, N * IDIM);
    k_cvt_w<<<(3 * IDIM * ODIM + TB - 1) / TB, TB, 0, stream>>>(Wl, Wr, skip_W, wb);

    int tilesPerBlock = 8;                                   // 8 waves * 16 rows = 128 rows/block
    int gemmBlocks = (N / 16 + tilesPerBlock - 1) / tilesPerBlock;
    k_gemm<<<gemmBlocks, TB, 0, stream>>>(xb, wb, skip_b, xl, xr, xsk, N);

    int edgeThreads = Etot * HEADS;
    int eb = (edgeThreads + TB - 1) / TB;
    k_edge_score  <<<eb, TB, 0, stream>>>(ei, xl, xr, att, score, smax, E, Etot);
    k_edge_exp    <<<eb, TB, 0, stream>>>(ei, score, smax, denom, E, Etot);
    k_edge_scatter<<<eb, TB, 0, stream>>>(ei, score, denom, xl, accum, E, Etot);

    int nb = (nNodeElems + TB - 1) / TB;
    k_gsum <<<nb, TB, 0, stream>>>(accum, xsk, conv_b, batch, gstat, cnt, N);
    k_gvar <<<nb, TB, 0, stream>>>(accum, xsk, conv_b, batch, bn_ms, sn_ms, gstat, cnt, N);
    k_final<<<nb, TB, 0, stream>>>(accum, xsk, conv_b, batch, bn_w, bn_b, bn_ms,
                                   sn_w, sn_b, sn_ms, gstat, cnt, (float*)d_out, N);
}
